// Bottleneck_quant_20572893348556
// MI455X (gfx1250) — matrix operand order repaired and checked by reading, not compiled, not run on hardware
//
#include <hip/hip_runtime.h>
#include <hip/hip_bf16.h>

typedef __attribute__((ext_vector_type(16))) _Float16 v16h;
typedef __attribute__((ext_vector_type(8)))  _Float16 v8h;
typedef __attribute__((ext_vector_type(8)))  float    v8f;
typedef __attribute__((ext_vector_type(4)))  float    v4f;

#define HW 784   // 28*28 = 49*16 -> 16-column chunks never straddle the boundary
#define NB 64    // batch

__device__ __forceinline__ v16h combine16(v8h lo, v8h hi) {
    v16h r;
#pragma unroll
    for (int i = 0; i < 8; ++i) { r[i] = lo[i]; r[i + 8] = hi[i]; }
    return r;
}

// ---------------------------------------------------------------------------
// Per-tensor symmetric 8-bit fake-quant -> f16 weights (w2 reordered tap-major)
// blockIdx.x: 0 = w1 (128x512), 1 = w2 (128x128x3x3), 2 = w3 (512x128)
// ---------------------------------------------------------------------------
__global__ __launch_bounds__(256) void quant_pack_kernel(
    const float* __restrict__ w1, const float* __restrict__ w2,
    const float* __restrict__ w3,
    _Float16* __restrict__ q1, _Float16* __restrict__ q2,
    _Float16* __restrict__ q3)
{
    __shared__ float red[256];
    const int which = blockIdx.x;
    const int tid = threadIdx.x;
    const float* src = (which == 0) ? w1 : (which == 1) ? w2 : w3;
    const int n = (which == 1) ? 128 * 128 * 9 : 128 * 512;

    float mx = 0.f;
    for (int i = tid; i < n; i += 256) mx = fmaxf(mx, fabsf(src[i]));
    red[tid] = mx;
    __syncthreads();
    for (int s = 128; s > 0; s >>= 1) {
        if (tid < s) red[tid] = fmaxf(red[tid], red[tid + s]);
        __syncthreads();
    }
    const float scale = red[0] / 127.0f;
    const float inv = 1.0f / scale;

    if (which == 1) {
        // OIHW (co, ci, ky, kx) -> [co][tap*128 + ci]
        for (int i = tid; i < n; i += 256) {
            float wq = rintf(src[i] * inv) * scale;   // round-half-even, matches jnp.round
            int co = i / 1152;
            int r = i - co * 1152;
            int ci = r / 9;
            int tap = r - ci * 9;
            q2[co * 1152 + tap * 128 + ci] = (_Float16)wq;
        }
    } else {
        _Float16* q = (which == 0) ? q1 : q3;
        for (int i = tid; i < n; i += 256) {
            q[i] = (_Float16)(rintf(src[i] * inv) * scale);
        }
    }
}

// ---------------------------------------------------------------------------
// Implicit-GEMM conv via V_WMMA_F32_16X16X32_F16.
//   MODE 0: 1x1 conv, f32 source (x)
//   MODE 1: 3x3 conv pad 1, f16 source, K = 9*CSRC (tap-major)
//   MODE 2: 1x1 conv, f16 source
// Block tile: 128(M) x 128(N) x 32(K); 8 waves in 4(M)x2(N), each wave 32x64
// = 2x4 WMMA frags = 8 WMMA per K-step.
// grid: (ceil(784/128)=7, 64 images, COUT/128)
// ---------------------------------------------------------------------------
template <int MODE, int KTOT, int COUT, int CSRC>
__global__ __launch_bounds__(256) void gemm_conv_kernel(
    const void* __restrict__ srcv, const _Float16* __restrict__ Aw,
    const float* __restrict__ bias, _Float16* __restrict__ dst)
{
    constexpr int BK = 32, PA = 40, PB = 40;   // pitches keep b128 LDS alignment
    __shared__ __attribute__((aligned(16))) _Float16 As[128 * PA];
    __shared__ __attribute__((aligned(16))) _Float16 Bs[128 * PB];

    const int tid = threadIdx.x;
    const int lane = tid & 31;
    const int wave = tid >> 5;
    const int wm = wave & 3;          // 4 waves along M (32 rows each)
    const int wn = wave >> 2;         // 2 waves along N (64 cols each)
    const int img = blockIdx.y;
    const int pbase = blockIdx.x * 128;
    const int m0g = blockIdx.z * 128;

    const float* __restrict__ srcF = (const float*)srcv;
    const _Float16* __restrict__ srcH = (const _Float16*)srcv;

    v8f acc[2][4] = {};

    constexpr int NK = KTOT / BK;
    for (int ks = 0; ks < NK; ++ks) {
        // ---- stage A tile (128x32 f16): one row half-chunk per thread ----
        {
            const int row = tid >> 1;
            const int cc = (tid & 1) * 16;
            const v8h* gp = (const v8h*)(Aw + (size_t)(m0g + row) * KTOT + ks * BK + cc);
            v8h lo = gp[0];
            v8h hi = gp[1];
            *(v8h*)(&As[row * PA + cc]) = lo;
            *(v8h*)(&As[row * PA + cc + 8]) = hi;
            if (ks + 1 < NK)
                __builtin_prefetch(Aw + (size_t)(m0g + row) * KTOT + (ks + 1) * BK + cc, 0, 0);
        }
        // ---- stage B tile (32K x 128N), stored transposed Bs[n][k] ----
        // Each thread: one K row, 16 consecutive columns. Since HW % 16 == 0,
        // the 16-chunk validity test is uniform -> no per-element EXEC guards.
        {
            const int k = tid >> 3;            // 0..31
            const int nc = (tid & 7) * 16;     // 0,16,..,112
            const int jb = pbase + nc;         // chunk fully valid iff jb < HW
            const int kg = ks * BK + k;
            _Float16 hv[16];
            if (jb < HW) {
                if (MODE == 0) {
                    const float* sp = srcF + ((size_t)img * CSRC + kg) * HW + jb;
#pragma unroll
                    for (int c = 0; c < 4; ++c) {
                        v4f f = *(const v4f*)(sp + c * 4);
#pragma unroll
                        for (int u = 0; u < 4; ++u) hv[c * 4 + u] = (_Float16)f[u];
                    }
                } else if (MODE == 2) {
                    const v8h* sp = (const v8h*)(srcH + ((size_t)img * CSRC + kg) * HW + jb);
                    v8h h0 = sp[0];
                    v8h h1 = sp[1];
#pragma unroll
                    for (int u = 0; u < 8; ++u) { hv[u] = h0[u]; hv[u + 8] = h1[u]; }
                } else {
                    // 3x3 halo gather: branch-free (clamped index + select)
                    const int tap = kg >> 7;       // kg / 128
                    const int ci = kg & 127;
                    const int ky = tap / 3 - 1;
                    const int kx = tap % 3 - 1;
                    const _Float16* sp = srcH + ((size_t)img * CSRC + ci) * HW;
#pragma unroll
                    for (int i = 0; i < 16; ++i) {
                        const int j = jb + i;              // j <= 783 when jb < HW
                        const int py = j / 28 + ky;
                        const int px = j % 28 + kx;
                        const bool ok = ((unsigned)py < 28u) & ((unsigned)px < 28u);
                        const int idx = ok ? (py * 28 + px) : 0;
                        _Float16 f = sp[idx];
                        hv[i] = ok ? f : (_Float16)0.f;
                    }
                }
            } else {
#pragma unroll
                for (int i = 0; i < 16; ++i) hv[i] = (_Float16)0.f;
            }
#pragma unroll
            for (int i = 0; i < 16; ++i) Bs[(nc + i) * PB + k] = hv[i];
        }
        __syncthreads();

        // ---- build fragments per documented wave32 layouts ----
        v16h afr[2], bfr[4];
        {
            const int arow = wm * 32 + (lane & 15);
            const int akb = (lane < 16) ? 0 : 8;   // K halves {akb..+7, akb+16..+23}
#pragma unroll
            for (int i = 0; i < 2; ++i) {
                const _Float16* ap = &As[(arow + i * 16) * PA + akb];
                afr[i] = combine16(*(const v8h*)ap, *(const v8h*)(ap + 16));
            }
            const int bn = wn * 64 + (lane & 15);
            const int bkb = (lane < 16) ? 0 : 8;   // K halves {bkb..+7, bkb+16..+23}
#pragma unroll
            for (int j = 0; j < 4; ++j) {
                const _Float16* bp = &Bs[(bn + j * 16) * PB + bkb];
                bfr[j] = combine16(*(const v8h*)bp, *(const v8h*)(bp + 16));
            }
        }
#pragma unroll
        for (int i = 0; i < 2; ++i)
#pragma unroll
            for (int j = 0; j < 4; ++j)
                acc[i][j] = __builtin_amdgcn_wmma_f32_16x16x32_f16(
                    false, afr[i], false, bfr[j], (short)0, acc[i][j], false, false);
        __syncthreads();
    }

    // ---- epilogue: + bias, store f16 pre-BN ----
    const int ncol0 = pbase + wn * 64 + (lane & 15);
    const int mrow_off = (lane < 16) ? 0 : 8;
#pragma unroll
    for (int i = 0; i < 2; ++i) {
#pragma unroll
        for (int r = 0; r < 8; ++r) {
            const int m = m0g + wm * 32 + i * 16 + mrow_off + r;
            const float bv = bias[m];
#pragma unroll
            for (int j = 0; j < 4; ++j) {
                const int col = ncol0 + j * 16;
                if (col < HW)
                    dst[((size_t)img * COUT + m) * HW + col] = (_Float16)(acc[i][j][r] + bv);
            }
        }
    }
}

// ---------------------------------------------------------------------------
// Batch-norm statistics: one block per channel -> fold into y' = a*y + b
// ---------------------------------------------------------------------------
__global__ __launch_bounds__(256) void bn_stats_kernel(
    const _Float16* __restrict__ y, const float* __restrict__ gamma,
    const float* __restrict__ beta, float* __restrict__ ascale,
    float* __restrict__ bshift, int C)
{
    const int c = blockIdx.x;
    const int tid = threadIdx.x;
    __shared__ float s1[256], s2[256];
    float sum = 0.f, ss = 0.f;
    for (int i = tid; i < NB * HW; i += 256) {
        const int img = i / HW;
        const int p = i - img * HW;
        const float v = (float)y[((size_t)img * C + c) * HW + p];
        sum += v;
        ss += v * v;
    }
    s1[tid] = sum;
    s2[tid] = ss;
    __syncthreads();
    for (int s = 128; s > 0; s >>= 1) {
        if (tid < s) { s1[tid] += s1[tid + s]; s2[tid] += s2[tid + s]; }
        __syncthreads();
    }
    if (tid == 0) {
        const float n = (float)(NB * HW);
        const float mean = s1[0] / n;
        const float var = s2[0] / n - mean * mean;   // biased, matches jnp.var
        const float a = gamma[c] * rsqrtf(var + 1e-5f);
        ascale[c] = a;
        bshift[c] = beta[c] - mean * a;
    }
}

__global__ __launch_bounds__(256) void bn_relu_kernel(
    const _Float16* __restrict__ y, const float* __restrict__ a,
    const float* __restrict__ b, _Float16* __restrict__ out, int C, int total)
{
    const int i = blockIdx.x * 256 + threadIdx.x;
    if (i < total) {
        const int c = (i / HW) % C;
        const float v = a[c] * (float)y[i] + b[c];
        out[i] = (_Float16)fmaxf(v, 0.f);
    }
}

__global__ __launch_bounds__(256) void bn_add_relu_kernel(
    const _Float16* __restrict__ y3, const float* __restrict__ x,
    const float* __restrict__ a, const float* __restrict__ b,
    float* __restrict__ out, int total)
{
    const int i = blockIdx.x * 256 + threadIdx.x;
    if (i < total) {
        const int c = (i / HW) % 512;
        const float v = a[c] * (float)y3[i] + b[c] + x[i];
        out[i] = fmaxf(v, 0.f);
    }
}

// ---------------------------------------------------------------------------
extern "C" void kernel_launch(void* const* d_in, const int* in_sizes, int n_in,
                              void* d_out, int out_size, void* d_ws, size_t ws_size,
                              hipStream_t stream)
{
    const float* x   = (const float*)d_in[0];
    const float* w1  = (const float*)d_in[1];
    const float* b1  = (const float*)d_in[2];
    const float* g1  = (const float*)d_in[3];
    const float* be1 = (const float*)d_in[4];
    const float* w2  = (const float*)d_in[5];
    const float* b2  = (const float*)d_in[6];
    const float* g2  = (const float*)d_in[7];
    const float* be2 = (const float*)d_in[8];
    const float* w3  = (const float*)d_in[9];
    const float* b3  = (const float*)d_in[10];
    const float* g3  = (const float*)d_in[11];
    const float* be3 = (const float*)d_in[12];
    float* out = (float*)d_out;

    char* ws = (char*)d_ws;
    size_t off = 0;
    auto alloc = [&](size_t bytes) -> void* {
        void* p = ws + off;
        off += (bytes + 255) & ~(size_t)255;
        return p;
    };
    _Float16* q1 = (_Float16*)alloc(128 * 512 * 2);
    _Float16* q2 = (_Float16*)alloc(128 * 1152 * 2);
    _Float16* q3 = (_Float16*)alloc(512 * 128 * 2);
    float* a1 = (float*)alloc(128 * 4);
    float* s1 = (float*)alloc(128 * 4);
    float* a2 = (float*)alloc(128 * 4);
    float* s2 = (float*)alloc(128 * 4);
    float* a3 = (float*)alloc(512 * 4);
    float* s3 = (float*)alloc(512 * 4);
    _Float16* y1pre  = (_Float16*)alloc((size_t)NB * 128 * HW * 2);
    _Float16* y1post = (_Float16*)alloc((size_t)NB * 128 * HW * 2);
    _Float16* y2pre  = (_Float16*)alloc((size_t)NB * 128 * HW * 2);
    _Float16* y2post = (_Float16*)alloc((size_t)NB * 128 * HW * 2);
    _Float16* y3pre  = (_Float16*)alloc((size_t)NB * 512 * HW * 2);

    const int tot128 = NB * 128 * HW;   // 6,422,528
    const int tot512 = NB * 512 * HW;   // 25,690,112

    quant_pack_kernel<<<3, 256, 0, stream>>>(w1, w2, w3, q1, q2, q3);

    dim3 gconv(7, NB, 1);
    gemm_conv_kernel<0, 512, 128, 512><<<gconv, 256, 0, stream>>>(x, q1, b1, y1pre);
    bn_stats_kernel<<<128, 256, 0, stream>>>(y1pre, g1, be1, a1, s1, 128);
    bn_relu_kernel<<<tot128 / 256, 256, 0, stream>>>(y1pre, a1, s1, y1post, 128, tot128);

    gemm_conv_kernel<1, 1152, 128, 128><<<gconv, 256, 0, stream>>>(y1post, q2, b2, y2pre);
    bn_stats_kernel<<<128, 256, 0, stream>>>(y2pre, g2, be2, a2, s2, 128);
    bn_relu_kernel<<<tot128 / 256, 256, 0, stream>>>(y2pre, a2, s2, y2post, 128, tot128);

    dim3 gconv3(7, NB, 4);
    gemm_conv_kernel<2, 128, 512, 128><<<gconv3, 256, 0, stream>>>(y2post, q3, b3, y3pre);
    bn_stats_kernel<<<512, 256, 0, stream>>>(y3pre, g3, be3, a3, s3, 512);
    bn_add_relu_kernel<<<tot512 / 256, 256, 0, stream>>>(y3pre, x, a3, s3, out, tot512);
}
